// Group_SSM_55465207661135
// MI455X (gfx1250) — matrix operand order repaired and checked by reading, not compiled, not run on hardware
//
#include <hip/hip_runtime.h>
#include <hip/hip_bf16.h>
#include <math.h>

// ---------------------------------------------------------------------------
// Group-SSM block for MI455X (gfx1250, wave32, WMMA).
//
// conv1d (conv_w 4096x4096x3) dominates: ~51.5 GFLOP. conv_w is (l_out,l_in,k)
// row-major, i.e. contiguous over K' = l_in*3 + k. Building the B-matrix with
// interleaved rows B'[l_in*3+t, col] = X[b, l_in, h+t-1] turns the 3-tap conv
// into ONE GEMM with K=12288 whose A-matrix is conv_w itself: f32 loaded
// contiguously and converted to bf16 in-register (no 100MB transposed copy,
// no extra 300MB HBM pass). bf16 WMMA, f32 accumulation.
// S6 collapses to an 8x8 quadratic form (h0 == 0); A input is unused.
// Workspace: ~21 MB.
// ---------------------------------------------------------------------------

typedef __bf16 v8bf  __attribute__((ext_vector_type(8)));
typedef __bf16 v16bf __attribute__((ext_vector_type(16)));
typedef float  v8f   __attribute__((ext_vector_type(8)));

union AV { v16bf v; struct { v8bf lo, hi; } p; };

#define NL   4096
#define KP   12288L            // K' = 4096 * 3 taps

// --- 1) S6 constants: M2[8][8], lin[8], s ----------------------------------
// cb = x^T M2 x + lin.x + s  with M2 = fc3_w^T fc2_w etc.
__global__ void k_s6_setup(const float* __restrict__ fc2_w,
                           const float* __restrict__ fc2_b,
                           const float* __restrict__ fc3_w,
                           const float* __restrict__ fc3_b,
                           float* __restrict__ s6c) {
  int t = threadIdx.x;
  if (t < 64) {
    int i = t >> 3, j = t & 7;
    float acc = 0.f;
    for (int n = 0; n < 128; ++n) acc += fc3_w[n * 8 + i] * fc2_w[n * 8 + j];
    s6c[t] = acc;
  } else if (t < 72) {
    int i = t - 64;
    float acc = 0.f;
    for (int n = 0; n < 128; ++n)
      acc += fc3_w[n * 8 + i] * fc2_b[n] + fc3_b[n] * fc2_w[n * 8 + i];
    s6c[64 + i] = acc;
  } else if (t == 72) {
    float acc = 0.f;
    for (int n = 0; n < 128; ++n) acc += fc2_b[n] * fc3_b[n];
    s6c[72] = acc;
  }
}

// --- 2) PEG(3x3 dw)+res -> RMSNorm -> inp_proj(16->32) -> interleaved B' ---
// B' column-major: Xi[col * 12288 + l*3 + t] = X[b, l, h + t - 1],
// col = b*32 + h. X value at channel j feeds (h,t) with h = j+1-t.
// Entries (h=0,t=0) and (h=31,t=2) are out-of-range taps -> zeroed.
__global__ void k_front(const float* __restrict__ in,
                        const float* __restrict__ peg_w,
                        const float* __restrict__ peg_b,
                        const float* __restrict__ norm_w,
                        const float* __restrict__ inp_w,
                        const float* __restrict__ inp_b,
                        __bf16* __restrict__ Xi) {
  int idx = blockIdx.x * blockDim.x + threadIdx.x;   // b*4096 + l
  int b = idx >> 12, l = idx & 4095;
  int h = l >> 6, w = l & 63;
  const float* ib = in + (long)b * 16 * NL;

  float x[16];
  float ssum = 0.f;
  for (int c = 0; c < 16; ++c) {
    float acc = peg_b[c];
#pragma unroll
    for (int dh = -1; dh <= 1; ++dh)
#pragma unroll
      for (int dw = -1; dw <= 1; ++dw) {
        int hh = h + dh, ww = w + dw;
        float v = 0.f;
        if (hh >= 0 && hh < 64 && ww >= 0 && ww < 64)
          v = ib[c * NL + hh * 64 + ww];
        acc += v * peg_w[c * 9 + (dh + 1) * 3 + (dw + 1)];
      }
    float xc = ib[c * NL + l] + acc;   // residual + PEG
    x[c] = xc;
    ssum += xc * xc;
  }
  float r = rsqrtf(ssum * (1.f / 16.f) + 1e-5f);
#pragma unroll
  for (int c = 0; c < 16; ++c) x[c] *= r * norm_w[c];

  long rowoff = (long)l * 3;
  long colbase = (long)(b * 32) * KP;
  for (int j = 0; j < 32; ++j) {
    float p = inp_b[j];
#pragma unroll
    for (int c = 0; c < 16; ++c) p += x[c] * inp_w[j * 16 + c];
    __bf16 pb = (__bf16)p;
#pragma unroll
    for (int t = 0; t < 3; ++t) {
      int ho = j + 1 - t;
      if (ho >= 0 && ho < 32)
        Xi[colbase + (long)ho * KP + rowoff + t] = pb;
    }
  }
  Xi[colbase + rowoff + 0] = (__bf16)0.f;                 // (h=0,  t=0)
  Xi[colbase + (long)31 * KP + rowoff + 2] = (__bf16)0.f; // (h=31, t=2)
}

// --- 3) GEMM: out[b,l_out,h] = sum_{K'} conv_w[l_out,K'] * B'[K', b*32+h] --
// K = 12288, A = conv_w (f32, contiguous) converted to bf16 in-register.
// Each wave: one 16-row M tile x four 16-col N tiles; 8 waves of a block
// share the same M tile's A rows (WGP-cache reuse).
__global__ void __launch_bounds__(256)
k_conv_gemm(const float* __restrict__ conv_w,
            const __bf16* __restrict__ Xi,
            float* __restrict__ convo) {
  int tid  = threadIdx.x;
  int lane = tid & 31;
  int wave = tid >> 5;
  int tileId = blockIdx.x * 8 + wave;   // 0..2047
  int cg = tileId & 7;                  // 8 groups of 64 columns (same mt/block)
  int mt = tileId >> 3;                 // 256 M tiles
  int m0 = mt * 16;
  int c0 = cg * 64;
  int ln = lane & 15;
  int hi = lane >> 4;                   // half-wave select

  long bcol[4];
#pragma unroll
  for (int s = 0; s < 4; ++s) bcol[s] = (long)(c0 + s * 16 + ln) * KP;

  // A-fragment (16x32 bf16): lane<16 holds K {0..15}, lane>=16 {16..31}
  const float* arow = conv_w + (long)(m0 + ln) * KP + (hi ? 16 : 0);
  // B-fragment (32x16 bf16): lane holds 16 consecutive K for its column
  long bhalf = hi ? 16 : 0;

  v8f acc[4] = {v8f{}, v8f{}, v8f{}, v8f{}};

#pragma unroll 1
  for (long kk = 0; kk < KP; kk += 32) {
    v8f alo = *(const v8f*)(arow + kk);
    v8f ahi = *(const v8f*)(arow + kk + 8);
    AV av;
#pragma unroll
    for (int i = 0; i < 8; ++i) {       // -> v_cvt_pk_bf16_f32 pairs
      av.v[i]     = (__bf16)alo[i];
      av.v[8 + i] = (__bf16)ahi[i];
    }
    long boff = kk + bhalf;
    v16bf b0 = *(const v16bf*)(Xi + bcol[0] + boff);
    v16bf b1 = *(const v16bf*)(Xi + bcol[1] + boff);
    v16bf b2 = *(const v16bf*)(Xi + bcol[2] + boff);
    v16bf b3 = *(const v16bf*)(Xi + bcol[3] + boff);
    acc[0] = __builtin_amdgcn_wmma_f32_16x16x32_bf16(false, av.v, false, b0,
                                                     (short)0, acc[0], false, false);
    acc[1] = __builtin_amdgcn_wmma_f32_16x16x32_bf16(false, av.v, false, b1,
                                                     (short)0, acc[1], false, false);
    acc[2] = __builtin_amdgcn_wmma_f32_16x16x32_bf16(false, av.v, false, b2,
                                                     (short)0, acc[2], false, false);
    acc[3] = __builtin_amdgcn_wmma_f32_16x16x32_bf16(false, av.v, false, b3,
                                                     (short)0, acc[3], false, false);
  }

  // C/D layout: VGPR r, lanes 0-15 -> M=r, lanes 16-31 -> M=r+8; N = lane&15
#pragma unroll
  for (int s = 0; s < 4; ++s) {
    int col = c0 + s * 16 + ln;
    int bb = col >> 5, hh = col & 31;
#pragma unroll
    for (int r = 0; r < 8; ++r) {
      int row = m0 + r + hi * 8;
      convo[(long)(bb * NL + row) * 32 + hh] = acc[s][r];
    }
  }
}

// --- 4) bias+SiLU -> S6 (quadratic form) -> SiLU -> shuffle -> out_proj+res -
__global__ void k_epilogue(const float* __restrict__ convo,
                           const float* __restrict__ conv_b,
                           const float* __restrict__ fc1_w,
                           const float* __restrict__ fc1_b,
                           const float* __restrict__ s6c,
                           const float* __restrict__ out_w,
                           const float* __restrict__ out_b,
                           const float* __restrict__ input,
                           float* __restrict__ out) {
  int idx = blockIdx.x * blockDim.x + threadIdx.x;   // b*4096 + l
  int b = idx >> 12, l = idx & 4095;
  float bias = conv_b[l];

  float xh[32];
  const float* cp = convo + (long)idx * 32;
#pragma unroll
  for (int h2 = 0; h2 < 32; ++h2) {
    float v = cp[h2] + bias;
    xh[h2] = v / (1.f + expf(-v));                   // SiLU
  }

  float y[32];
  for (int g = 0; g < 4; ++g) {
    const float* xc = xh + g * 8;
    float cb = s6c[72];
#pragma unroll
    for (int i = 0; i < 8; ++i) {
      float t = s6c[64 + i];
#pragma unroll
      for (int j = 0; j < 8; ++j) t += s6c[i * 8 + j] * xc[j];
      cb += xc[i] * t;
    }
#pragma unroll
    for (int j = 0; j < 8; ++j) {
      float d = fc1_b[j];
#pragma unroll
      for (int i = 0; i < 8; ++i) d += fc1_w[j * 8 + i] * xc[i];
      float sp = (d > 20.f) ? d : log1pf(expf(d));   // softplus
      float v = xc[j] * sp * cb;
      y[g * 8 + j] = v / (1.f + expf(-v));           // SiLU
    }
  }

  // ChannelShuffle(G=4): shuf[i*4+g] = y[g*8+i]; then out_proj 32->16 + residual
  const float* ib = input + (long)b * 16 * NL;
  for (int c = 0; c < 16; ++c) {
    float acc = out_b[c];
#pragma unroll
    for (int m = 0; m < 32; ++m) {
      int i = m >> 2, g = m & 3;
      acc += y[g * 8 + i] * out_w[c * 32 + m];
    }
    out[(long)(b * 16 + c) * NL + l] = acc + ib[c * NL + l];
  }
}

extern "C" void kernel_launch(void* const* d_in, const int* in_sizes, int n_in,
                              void* d_out, int out_size, void* d_ws, size_t ws_size,
                              hipStream_t stream) {
  const float* input  = (const float*)d_in[0];
  const float* peg_w  = (const float*)d_in[1];
  const float* peg_b  = (const float*)d_in[2];
  const float* norm_w = (const float*)d_in[3];
  const float* inp_w  = (const float*)d_in[4];
  const float* inp_b  = (const float*)d_in[5];
  const float* conv_w = (const float*)d_in[6];
  const float* conv_b = (const float*)d_in[7];
  const float* fc1_w  = (const float*)d_in[8];
  const float* fc1_b  = (const float*)d_in[9];
  const float* fc2_w  = (const float*)d_in[10];
  const float* fc2_b  = (const float*)d_in[11];
  const float* fc3_w  = (const float*)d_in[12];
  const float* fc3_b  = (const float*)d_in[13];
  // d_in[14] = A : unused (h0 == 0 collapses the recurrence; A cancels)
  const float* out_w  = (const float*)d_in[15];
  const float* out_b  = (const float*)d_in[16];
  float* out = (float*)d_out;

  // Workspace layout (~21 MB total)
  char* ws = (char*)d_ws;
  __bf16* Xi   = (__bf16*)ws;                 // 512 * 12288 * 2 = 12,582,912 B
  float* convo = (float*)(ws + 12582912);     // 16*4096*32*4    =  8,388,608 B
  float* s6c   = (float*)(ws + 20971520);     // 73 floats

  k_s6_setup<<<1, 128, 0, stream>>>(fc2_w, fc2_b, fc3_w, fc3_b, s6c);
  k_front<<<256, 256, 0, stream>>>(input, peg_w, peg_b, norm_w, inp_w, inp_b, Xi);
  k_conv_gemm<<<256, 256, 0, stream>>>(conv_w, Xi, convo);
  k_epilogue<<<256, 256, 0, stream>>>(convo, conv_b, fc1_w, fc1_b, s6c,
                                      out_w, out_b, input, out);
}
